// TGN_e2n_40166534152233
// MI455X (gfx1250) — matrix operand order repaired and checked by reading, not compiled, not run on hardware
//
#include <hip/hip_runtime.h>
#include <hip/hip_bf16.h>

// MI455X (gfx1250) fused temporal-graph attention.
// Compute-bound (~34 GFLOP vs ~12MB traffic @23.3TB/s) -> all matmuls via
// v_wmma_f32_16x16x32_f16. Weights pre-converted to f16 once so every
// B-fragment is a contiguous 32B v16h load.

typedef __attribute__((ext_vector_type(16))) _Float16 v16h;
typedef __attribute__((ext_vector_type(8)))  _Float16 v8h;
typedef __attribute__((ext_vector_type(8)))  float    v8f;

#define DD 128          // embed dim
#define SQRT_INV_D 0.08838834764831845f   // sqrt(1/128) == 1/sqrt(128)

__device__ __forceinline__ v8f wmma_f16(v16h a, v16h b, v8f c) {
    return __builtin_amdgcn_wmma_f32_16x16x32_f16(
        /*neg_a=*/false, a, /*neg_b=*/false, b,
        /*c_mod=*/(short)0, c, /*reuse_a=*/false, /*reuse_b=*/false);
}

// A-fragment (16x32 f16, M x K). lane m = lane&15, half h = lane>>4.
// Contiguous layout (matches B): half0 slots hold K {0..15}, half1 holds K {16..31}.
__device__ __forceinline__ v16h load_afrag(const _Float16* base, int m, int h,
                                           int kbase, int stride) {
    v16h a;
#pragma unroll
    for (int j = 0; j < 8; ++j) a[j]     = base[m * stride + kbase + h * 16 + j];
#pragma unroll
    for (int j = 0; j < 8; ++j) a[8 + j] = base[m * stride + kbase + h * 16 + 8 + j];
    return a;
}

// B-fragment (32x16 f16, K x N): column n = row `nrow` of a row-major f16
// matrix (B = W^T). lane n holds K {h*16 .. h*16+15} -> one 32B vector load.
__device__ __forceinline__ v16h load_bfrag_h(const _Float16* Wh, int nrow,
                                             int kbase, int h) {
    return *(const v16h*)(Wh + nrow * DD + kbase + h * 16);
}

// ---------------------------------------------------------------------------
// Kernel 0: one-time f32 -> f16 conversion of the eight 128x128 weights.
// ---------------------------------------------------------------------------
__global__ __launch_bounds__(256) void cvt_weights(
    const float* W0, const float* W1, const float* W2, const float* W3,
    const float* W4, const float* W5, const float* W6, const float* W7,
    _Float16* __restrict__ dst)
{
    const int mat = blockIdx.y;
    const int i   = blockIdx.x * 256 + threadIdx.x;   // 64 x-blocks * 256 = 16384
    const float* W;
    switch (mat) {
        case 0: W = W0; break; case 1: W = W1; break;
        case 2: W = W2; break; case 3: W = W3; break;
        case 4: W = W4; break; case 5: W = W5; break;
        case 6: W = W6; break; default: W = W7; break;
    }
    dst[mat * (DD * DD) + i] = (_Float16)W[i];
}

// ---------------------------------------------------------------------------
// Kernel 1: harmonic-encode keys + project with Wk/Wv for both heads.
// One block = 16 keys. kp row-major [Spad,128]; V transposed [128,Spad].
// proj is a compile-time constant (unrolled) -> straight-line stores.
// ---------------------------------------------------------------------------
__global__ __launch_bounds__(256) void encode_kv(
    const float* __restrict__ e_ts, const int* __restrict__ e_exp,
    const int* __restrict__ target, const float* __restrict__ freq,
    const _Float16* Wk0, const float* bk0, const _Float16* Wv0, const float* bv0,
    const _Float16* Wk1, const float* bk1, const _Float16* Wv1, const float* bv1,
    _Float16* __restrict__ kp0, _Float16* __restrict__ kp1,
    _Float16* __restrict__ vT0, _Float16* __restrict__ vT1,
    float* __restrict__ ts_pad, int* __restrict__ flagv, int S, int Spad)
{
    __shared__ _Float16 lphi[16 * DD];
    const int tid = threadIdx.x;
    const int s0  = blockIdx.x * 16;
    const int tgt = target[0];

#pragma unroll
    for (int it = 0; it < 8; ++it) {
        int idx = tid + it * 256;
        int row = idx >> 7, col = idx & 127;
        int s = s0 + row;
        float ts = 0.f;
        if (s >= 2 && s < S) ts = e_ts[s - 2];
        float ph = ts * freq[col & 63];
        float v  = (col < 64) ? __cosf(ph) : __sinf(ph);
        lphi[idx] = (_Float16)(SQRT_INV_D * v);
    }
    if (tid < 16) {
        int s = s0 + tid;
        float ts; int fl;
        if (s >= S)      { ts = 3.0e38f; fl = 0; }      // padding: always masked
        else if (s == 0) { ts = 0.f;     fl = 2; }      // sentinel exp=-1 -> neighbor
        else if (s == 1) { ts = 0.f;     fl = 1; }      // sentinel exp=target -> self
        else { ts = e_ts[s - 2]; fl = (e_exp[s - 2] == tgt) ? 1 : 2; }
        ts_pad[s] = ts; flagv[s] = fl;
    }
    __syncthreads();

    const int lane = tid & 31, wid = tid >> 5;
    const int n16 = lane & 15, h = lane >> 4;
    const int nt  = wid;                                // this wave's 16-col tile

    // A-fragments are shared by all four projections: hoist them.
    v16h a[4];
#pragma unroll
    for (int c = 0; c < 4; ++c) a[c] = load_afrag(lphi, n16, h, 32 * c, DD);

#pragma unroll
    for (int proj = 0; proj < 4; ++proj) {              // compile-time constant
        const _Float16* W; const float* bias;
        switch (proj) {
            case 0:  W = Wk0; bias = bk0; break;
            case 1:  W = Wv0; bias = bv0; break;
            case 2:  W = Wk1; bias = bk1; break;
            default: W = Wv1; bias = bv1; break;
        }
        // preload all B-fragments so loads clause & overlap the WMMA chain
        v16h b[4];
#pragma unroll
        for (int c = 0; c < 4; ++c) b[c] = load_bfrag_h(W, 16 * nt + n16, 32 * c, h);
        v8f acc = {};
#pragma unroll
        for (int c = 0; c < 4; ++c) acc = wmma_f16(a[c], b[c], acc);

        float bb = bias[16 * nt + n16];
        if (proj == 0 || proj == 2) {
            _Float16* kp = (proj == 0) ? kp0 : kp1;     // row-major, strided rows
#pragma unroll
            for (int r = 0; r < 8; ++r)
                kp[(s0 + 8 * h + r) * DD + 16 * nt + n16] = (_Float16)(acc[r] + bb);
        } else {
            _Float16* vT = (proj == 1) ? vT0 : vT1;     // transposed: r's contiguous
            v8h pk;
#pragma unroll
            for (int r = 0; r < 8; ++r) pk[r] = (_Float16)(acc[r] + bb);
            *(v8h*)(vT + (size_t)(16 * nt + n16) * Spad + s0 + 8 * h) = pk;
        }
    }
}

// ---------------------------------------------------------------------------
// Kernel 2: harmonic-encode queries + project with Wq for both heads.
// ---------------------------------------------------------------------------
__global__ __launch_bounds__(256) void encode_q(
    const float* __restrict__ tq, const float* __restrict__ freq,
    const _Float16* Wq0, const float* bq0, const _Float16* Wq1, const float* bq1,
    _Float16* __restrict__ qp0, _Float16* __restrict__ qp1)
{
    __shared__ _Float16 lphi[16 * DD];
    const int tid = threadIdx.x;
    const int s0  = blockIdx.x * 16;

#pragma unroll
    for (int it = 0; it < 8; ++it) {
        int idx = tid + it * 256;
        int row = idx >> 7, col = idx & 127;
        float ts = tq[s0 + row];
        float ph = ts * freq[col & 63];
        float v  = (col < 64) ? __cosf(ph) : __sinf(ph);
        lphi[idx] = (_Float16)(SQRT_INV_D * v);
    }
    __syncthreads();

    const int lane = tid & 31, wid = tid >> 5;
    const int n16 = lane & 15, h = lane >> 4;
    const int nt  = wid;

    v16h a[4];
#pragma unroll
    for (int c = 0; c < 4; ++c) a[c] = load_afrag(lphi, n16, h, 32 * c, DD);

#pragma unroll
    for (int proj = 0; proj < 2; ++proj) {              // compile-time constant
        const _Float16* W  = proj ? Wq1 : Wq0;
        const float* bias  = proj ? bq1 : bq0;
        _Float16*    out   = proj ? qp1 : qp0;
        v16h b[4];
#pragma unroll
        for (int c = 0; c < 4; ++c) b[c] = load_bfrag_h(W, 16 * nt + n16, 32 * c, h);
        v8f acc = {};
#pragma unroll
        for (int c = 0; c < 4; ++c) acc = wmma_f16(a[c], b[c], acc);
        float bb = bias[16 * nt + n16];
#pragma unroll
        for (int r = 0; r < 8; ++r)
            out[(s0 + 8 * h + r) * DD + 16 * nt + n16] = (_Float16)(acc[r] + bb);
    }
}

// ---------------------------------------------------------------------------
// Kernel 3: flash attention. blockIdx.y = head; 16 query rows per wave,
// 128 rows per 256-thread block. 64-key chunks: 32 WMMAs per chunk, one
// shuffle-reduction pass per chunk (1 shuffle/key amortized).
// ---------------------------------------------------------------------------
__global__ __launch_bounds__(256) void flash_attn(
    const float* __restrict__ tq, const float* __restrict__ ts_pad,
    const int* __restrict__ flagv,
    const _Float16* __restrict__ qp_s, const _Float16* __restrict__ qp_n,
    const _Float16* __restrict__ kp_s, const _Float16* __restrict__ kp_n,
    const _Float16* __restrict__ vT_s, const _Float16* __restrict__ vT_n,
    const _Float16* Wo_s, const float* bo_s,
    const _Float16* Wo_n, const float* bo_n,
    float* __restrict__ out_atten, int Spad)
{
    __shared__ _Float16 lbuf[8][16 * DD];    // per-wave staging (P / O reshape)

    const int tid  = threadIdx.x;
    const int lane = tid & 31, wid = tid >> 5;
    const int n16  = lane & 15, h = lane >> 4;
    const int head = blockIdx.y;

    const _Float16* qp = head ? qp_n : qp_s;
    const _Float16* kp = head ? kp_n : kp_s;
    const _Float16* vT = head ? vT_n : vT_s;
    const _Float16* Wo = head ? Wo_n : Wo_s;
    const float*    bo = head ? bo_n : bo_s;
    const int headflag = head + 1;           // 1=self-allowed, 2=neighbor-allowed
    const int colOff   = head * DD;

    const int row0 = blockIdx.x * 128 + wid * 16;

    v16h qa[4];
#pragma unroll
    for (int c = 0; c < 4; ++c)
        qa[c] = load_afrag(qp + (size_t)row0 * DD, n16, h, 32 * c, DD);

    v8f   O[8] = {};                         // 16x128 f32 accumulator
    float rm[8], rl[8], tqv[8];
#pragma unroll
    for (int r = 0; r < 8; ++r) {
        rm[r] = -3.0e38f; rl[r] = 0.f;
        tqv[r] = tq[row0 + 8 * h + r];
    }

    _Float16* P = &lbuf[wid][0];             // 16x64 P staging tile

    for (int kb = 0; kb < Spad; kb += 64) {
        if (kb + 64 < Spad)
            __builtin_prefetch(kp + (size_t)(kb + 64) * DD, 0, 0);  // global_prefetch_b8

        // --- scores: S = Q @ K^T, four 16x16 N-subtiles ---
        v8f sc[4];
#pragma unroll
        for (int sub = 0; sub < 4; ++sub) {
            const _Float16* krow = kp + (size_t)(kb + 16 * sub + n16) * DD;
            v16h b[4];
#pragma unroll
            for (int c = 0; c < 4; ++c) b[c] = *(const v16h*)(krow + 32 * c + h * 16);
            v8f s = {};
#pragma unroll
            for (int c = 0; c < 4; ++c) s = wmma_f16(qa[c], b[c], s);
            sc[sub] = s;
        }
        float tk[4]; bool ok[4];
#pragma unroll
        for (int sub = 0; sub < 4; ++sub) {
            tk[sub] = ts_pad[kb + 16 * sub + n16];
            ok[sub] = (flagv[kb + 16 * sub + n16] == headflag);
        }

        // --- online softmax, per row r (row m = 8h+r, cols = lanes) ---
#pragma unroll
        for (int r = 0; r < 8; ++r) {
            float sv[4];
#pragma unroll
            for (int sub = 0; sub < 4; ++sub) {
                float v = sc[sub][r] * SQRT_INV_D;
                if (!(ok[sub] && tk[sub] <= tqv[r])) v = -1.0e30f;
                sv[sub] = v;
            }
            float mx = fmaxf(fmaxf(sv[0], sv[1]), fmaxf(sv[2], sv[3]));
            mx = fmaxf(mx, __shfl_xor(mx, 1, 32));
            mx = fmaxf(mx, __shfl_xor(mx, 2, 32));
            mx = fmaxf(mx, __shfl_xor(mx, 4, 32));
            mx = fmaxf(mx, __shfl_xor(mx, 8, 32));
            bool valid = mx > -5.0e29f;      // whole chunk masked for this row?
            float newm = valid ? fmaxf(rm[r], mx) : rm[r];
            float f = __expf(rm[r] - newm);  // ==1 when nothing changed
            float e[4], ps = 0.f;
#pragma unroll
            for (int sub = 0; sub < 4; ++sub) {
                e[sub] = valid ? __expf(sv[sub] - newm) : 0.f;
                ps += e[sub];
            }
            ps += __shfl_xor(ps, 1, 32);
            ps += __shfl_xor(ps, 2, 32);
            ps += __shfl_xor(ps, 4, 32);
            ps += __shfl_xor(ps, 8, 32);
            rl[r] = rl[r] * f + ps;
            rm[r] = newm;
#pragma unroll
            for (int dt = 0; dt < 8; ++dt) O[dt][r] *= f;
#pragma unroll
            for (int sub = 0; sub < 4; ++sub)
                P[(8 * h + r) * 64 + 16 * sub + n16] = (_Float16)e[sub];
        }
        // reshape to A-layout (in-wave DS ops are in-order; no barrier needed)
        v16h pa0 = load_afrag(P, n16, h, 0,  64);
        v16h pa1 = load_afrag(P, n16, h, 32, 64);

        // --- O += P @ V (K = 64 keys), 8 d-tiles x 2 k-chunks ---
#pragma unroll
        for (int dt = 0; dt < 8; ++dt) {
            const _Float16* vcol = vT + (size_t)(16 * dt + n16) * Spad + kb;
            v16h b0 = *(const v16h*)(vcol + h * 16);
            v16h b1 = *(const v16h*)(vcol + 32 + h * 16);
            O[dt] = wmma_f16(pa0, b0, O[dt]);
            O[dt] = wmma_f16(pa1, b1, O[dt]);
        }
    }

    // --- normalize and stage O for the output projection ---
    float inv[8];
#pragma unroll
    for (int r = 0; r < 8; ++r) inv[r] = (rl[r] > 0.f) ? 1.f / rl[r] : 0.f;
    _Float16* lO = &lbuf[wid][0];
#pragma unroll
    for (int dt = 0; dt < 8; ++dt)
#pragma unroll
        for (int r = 0; r < 8; ++r)
            lO[(8 * h + r) * DD + 16 * dt + n16] = (_Float16)(O[dt][r] * inv[r]);

    // --- out = O @ Wo^T + bo ---
#pragma unroll
    for (int nt = 0; nt < 8; ++nt) {
        v16h b[4];
#pragma unroll
        for (int c = 0; c < 4; ++c) b[c] = load_bfrag_h(Wo, 16 * nt + n16, 32 * c, h);
        v8f acc = {};
#pragma unroll
        for (int c = 0; c < 4; ++c) {
            v16h a = load_afrag(lO, n16, h, 32 * c, DD);
            acc = wmma_f16(a, b[c], acc);
        }
        float bb = bo[16 * nt + n16];
#pragma unroll
        for (int r = 0; r < 8; ++r)
            out_atten[(size_t)(row0 + 8 * h + r) * 256 + colOff + 16 * nt + n16]
                = acc[r] + bb;
    }
}

// ---------------------------------------------------------------------------
// Kernel 4: lam = atten @ W_lin^T + b_lin ; clamped softplus with phi.
// ---------------------------------------------------------------------------
__global__ void lambda_head(const float* __restrict__ atten,
                            const float* __restrict__ W_lin,
                            const float* __restrict__ b_lin,
                            const float* __restrict__ phiArr,
                            float* __restrict__ lam, int T)
{
    int t = blockIdx.x * blockDim.x + threadIdx.x;
    if (t >= T) return;
    const float* row = atten + (size_t)t * 256;
    float acc = b_lin[0];
#pragma unroll 8
    for (int j = 0; j < 256; ++j) acc += row[j] * W_lin[j];
    float ph = phiArr[0];
    float x = acc * ph;
    x = fminf(x, 20.f);
    lam[t] = log1pf(__expf(x)) / ph;
}

// ---------------------------------------------------------------------------
extern "C" void kernel_launch(void* const* d_in, const int* in_sizes, int n_in,
                              void* d_out, int out_size, void* d_ws, size_t ws_size,
                              hipStream_t stream) {
    const float* t        = (const float*)d_in[0];
    const float* e_ts     = (const float*)d_in[1];
    const int*   e_exp    = (const int*)  d_in[2];
    const int*   target   = (const int*)  d_in[3];
    const float* freq     = (const float*)d_in[4];
    const float* phi      = (const float*)d_in[5];
    const float* W_lin    = (const float*)d_in[6];
    const float* b_lin    = (const float*)d_in[7];
    const float* Wq_s = (const float*)d_in[8],  *Wk_s = (const float*)d_in[9];
    const float* Wv_s = (const float*)d_in[10], *Wo_s = (const float*)d_in[11];
    const float* bq_s = (const float*)d_in[12], *bk_s = (const float*)d_in[13];
    const float* bv_s = (const float*)d_in[14], *bo_s = (const float*)d_in[15];
    const float* Wq_n = (const float*)d_in[16], *Wk_n = (const float*)d_in[17];
    const float* Wv_n = (const float*)d_in[18], *Wo_n = (const float*)d_in[19];
    const float* bq_n = (const float*)d_in[20], *bk_n = (const float*)d_in[21];
    const float* bv_n = (const float*)d_in[22], *bo_n = (const float*)d_in[23];

    const int T    = in_sizes[0];
    const int L    = in_sizes[1];
    const int S    = L + 2;
    const int Spad = (S + 63) & ~63;

    // workspace layout
    char* p = (char*)d_ws;
    _Float16* wh   = (_Float16*)p; p += (size_t)8 * DD * DD * sizeof(_Float16);
    _Float16* kp_s = (_Float16*)p; p += (size_t)Spad * DD * sizeof(_Float16);
    _Float16* kp_n = (_Float16*)p; p += (size_t)Spad * DD * sizeof(_Float16);
    _Float16* vT_s = (_Float16*)p; p += (size_t)DD * Spad * sizeof(_Float16);
    _Float16* vT_n = (_Float16*)p; p += (size_t)DD * Spad * sizeof(_Float16);
    _Float16* qp_s = (_Float16*)p; p += (size_t)T * DD * sizeof(_Float16);
    _Float16* qp_n = (_Float16*)p; p += (size_t)T * DD * sizeof(_Float16);
    float*    ts_pad = (float*)p;  p += (size_t)Spad * sizeof(float);
    int*      flagv  = (int*)p;    p += (size_t)Spad * sizeof(int);

    const int WSZ = DD * DD;
    _Float16* whQ_s = wh + 0 * WSZ, *whK_s = wh + 1 * WSZ;
    _Float16* whV_s = wh + 2 * WSZ, *whO_s = wh + 3 * WSZ;
    _Float16* whQ_n = wh + 4 * WSZ, *whK_n = wh + 5 * WSZ;
    _Float16* whV_n = wh + 6 * WSZ, *whO_n = wh + 7 * WSZ;

    float* out_lam   = (float*)d_out;        // (T,)
    float* out_atten = (float*)d_out + T;    // (T, 256)

    dim3 gw(WSZ / 256, 8);
    cvt_weights<<<gw, 256, 0, stream>>>(
        Wq_s, Wk_s, Wv_s, Wo_s, Wq_n, Wk_n, Wv_n, Wo_n, wh);

    encode_kv<<<Spad / 16, 256, 0, stream>>>(
        e_ts, e_exp, target, freq,
        whK_s, bk_s, whV_s, bv_s, whK_n, bk_n, whV_n, bv_n,
        kp_s, kp_n, vT_s, vT_n, ts_pad, flagv, S, Spad);

    encode_q<<<T / 16, 256, 0, stream>>>(
        t, freq, whQ_s, bq_s, whQ_n, bq_n, qp_s, qp_n);

    dim3 g2(T / 128, 2);
    flash_attn<<<g2, 256, 0, stream>>>(
        t, ts_pad, flagv, qp_s, qp_n, kp_s, kp_n, vT_s, vT_n,
        whO_s, bo_s, whO_n, bo_n, out_atten, Spad);

    lambda_head<<<(T + 255) / 256, 256, 0, stream>>>(
        out_atten, W_lin, b_lin, phi, out_lam, T);
}
